// TextClassifier_82094004896188
// MI455X (gfx1250) — matrix operand order repaired and checked by reading, not compiled, not run on hardware
//
#include <hip/hip_runtime.h>
#include <cstdint>
#include <cstddef>

// Problem constants (from the reference)
#define VOCAB 50000
#define EMB   100
#define HID   512
#define BB    512
#define TT    1024
#define KP    128      // embedding K padded 100 -> 128 for WMMA K-tiling
#define H3    1536     // 3*HID gate dimension

typedef __attribute__((ext_vector_type(16))) __bf16 v16bf;
typedef __attribute__((ext_vector_type(8)))  float  v8f;

#define MTILE   32                  // batch rows per workgroup
#define NBLK    (BB / MTILE)        // 16 persistent workgroups
#define WAVES   8
#define THREADS (WAVES * 32)
#define TPW     8                   // (MTILE/16)*(HID/16)/WAVES tiles per wave

// LDS carve-out (dynamic shared):
//   [0,      65536) : h double buffer, 2 x 32 x 512 bf16
//   [65536,  81920) : Xb stage double buffer, 2 x 32 x 128 bf16
#define SMEM_BYTES (65536 + 16384)

// ---------------------------------------------------------------------------
// CDNA5 async global->LDS copy (ASYNCcnt-tracked), 16B granule.
// Builtin signature (from toolchain diagnostic): pointers are
// vector_size(16)-int typed in AS1 (global src) / AS3 (LDS dst).
// ---------------------------------------------------------------------------
typedef int v4i_t __attribute__((vector_size(16)));

static __device__ __forceinline__ void async_copy16(const __bf16* gsrc, __bf16* ldst) {
#if __has_builtin(__builtin_amdgcn_global_load_async_to_lds_b128)
  __builtin_amdgcn_global_load_async_to_lds_b128(
      (__attribute__((address_space(1))) v4i_t*)(void*)gsrc,
      (__attribute__((address_space(3))) v4i_t*)(void*)ldst, 0, 0);
#else
  asm volatile("global_load_async_to_lds_b128 %0, %1, off"
               :: "v"((unsigned)(uintptr_t)ldst), "v"(gsrc) : "memory");
#endif
}

static __device__ __forceinline__ void wait_async0() {
#if __has_builtin(__builtin_amdgcn_s_wait_asynccnt)
  __builtin_amdgcn_s_wait_asynccnt(0);
#else
  asm volatile("s_wait_asynccnt 0x0" ::: "memory");
#endif
}

// Stage one 8 KB timestep slab (32 rows x 128 bf16, contiguous in global)
// into LDS: 512 x 16B chunks, 2 per thread.
static __device__ __forceinline__ void stage_slab(const __bf16* gslab, __bf16* lslab, int tid) {
#pragma unroll
  for (int j = 0; j < 2; ++j) {
    int c = tid * 2 + j;               // chunk id 0..511
    async_copy16(gslab + c * 8, lslab + c * 8);
  }
}

// Load a 16-element bf16 fragment as two 16B chunks (one lane's share).
template <typename P>
static __device__ __forceinline__ v16bf load_frag(P p0, P p1) {
  union { v16bf v; uint4 q[2]; } u;
  u.q[0] = *(const uint4*)p0;
  u.q[1] = *(const uint4*)p1;
  return u.v;
}

// ---------------------------------------------------------------------------
// Kernel 1: embedding gather -> bf16, K padded to 128, laid out time-major
// Xb[t][b][k] so each timestep is a contiguous slab streamed once.
// ---------------------------------------------------------------------------
__global__ void embed_pack_kernel(const int* __restrict__ x_in,
                                  const float* __restrict__ emb,
                                  __bf16* __restrict__ Xb) {
  long gid = (long)blockIdx.x * blockDim.x + threadIdx.x;  // B*T*KP threads
  long row = gid >> 7;                                     // t*B + b
  int  e   = (int)(gid & (KP - 1));
  if (row >= (long)BB * TT) return;
  int t = (int)(row / BB);
  int b = (int)(row - (long)t * BB);
  int tok = x_in[(long)b * TT + t];
  float v = (e < EMB) ? emb[(long)tok * EMB + e] : 0.0f;
  Xb[gid] = (__bf16)v;
}

// ---------------------------------------------------------------------------
// Kernel 2: pack W_ih (K padded) and W_hh to bf16 [N=3H][K] row-major.
// Both stay L2-resident for the whole recurrent kernel (1.9 MB total).
// ---------------------------------------------------------------------------
__global__ void pack_weights_kernel(const float* __restrict__ W_ih,
                                    const float* __restrict__ W_hh,
                                    __bf16* __restrict__ Wihb,
                                    __bf16* __restrict__ Whhb) {
  int stride = gridDim.x * blockDim.x;
  for (int i = blockIdx.x * blockDim.x + threadIdx.x; i < H3 * KP; i += stride) {
    int n = i >> 7, k = i & (KP - 1);
    Wihb[i] = (k < EMB) ? (__bf16)W_ih[n * EMB + k] : (__bf16)0.0f;
  }
  for (int i = blockIdx.x * blockDim.x + threadIdx.x; i < H3 * HID; i += stride)
    Whhb[i] = (__bf16)W_hh[i];
}

// ---------------------------------------------------------------------------
// Kernel 3: persistent GRU. h tile double-buffered in LDS, Xb slab
// double-buffered via async global->LDS DMA overlapped with each step's
// WMMA work. Gates via v_wmma_f32_16x16x32_bf16, fp32 accumulators.
// ---------------------------------------------------------------------------
__global__ __launch_bounds__(THREADS)
void gru_persistent_kernel(const __bf16* __restrict__ Xb,
                           const __bf16* __restrict__ Wihb,
                           const __bf16* __restrict__ Whhb,
                           const float* __restrict__ b_ih,
                           const float* __restrict__ b_hh,
                           const float* __restrict__ W_fc,
                           const float* __restrict__ b_fc,
                           float* __restrict__ out) {
  extern __shared__ __align__(16) unsigned char smem[];
  __bf16* hbase = (__bf16*)smem;                 // [2][MTILE][HID]
  __bf16* xbase = (__bf16*)(smem + 65536);       // [2][MTILE][KP]

  const int tid  = threadIdx.x;
  const int lane = tid & 31;
  const int wave = tid >> 5;
  const int blockBase = blockIdx.x * MTILE;

  // h0 = 0
  for (int i = tid; i < MTILE * HID; i += THREADS)
    hbase[i] = (__bf16)0.0f;

  // Prologue: stage slab for t=0, wait, publish.
  stage_slab(Xb + (long)blockBase * KP, xbase, tid);
  wait_async0();
  __syncthreads();

  const int colLocal = lane & 15;    // N within 16x16 tile (B/C/D lane layout)
  const int hl       = lane >> 4;    // which half of the wave

  // Per-tile constants (tile assignment fixed across timesteps).
  int m0s[TPW], n0s[TPW];
  float brv[TPW], bzv[TPW], binv[TPW], bhnv[TPW];
#pragma unroll
  for (int i = 0; i < TPW; ++i) {
    int tile = i * WAVES + wave;          // 0..63
    int m0 = (tile & 1) * 16;
    int n0 = (tile >> 1) * 16;
    m0s[i] = m0; n0s[i] = n0;
    int c = n0 + colLocal;
    brv[i]  = b_ih[c]           + b_hh[c];            // r: biases fuse
    bzv[i]  = b_ih[HID + c]     + b_hh[HID + c];      // z: biases fuse
    binv[i] = b_ih[2 * HID + c];                      // n: input part
    bhnv[i] = b_hh[2 * HID + c];                      // n: hidden part (scaled by r)
  }

  int cur = 0;
  for (int t = 0; t < TT; ++t) {
    const int xc = t & 1;
    // Kick off DMA for the next timestep's slab; completes during this step.
    if (t + 1 < TT)
      stage_slab(Xb + ((long)(t + 1) * BB + blockBase) * KP,
                 xbase + (xc ^ 1) * (MTILE * KP), tid);
    if ((t + 2 < TT) && (lane == 0))
      __builtin_prefetch(Xb + ((long)(t + 2) * BB + blockBase) * KP, 0, 0);

    const __bf16* xcur = xbase + xc * (MTILE * KP);
    const __bf16* hcur = hbase + cur * (MTILE * HID);
    __bf16*       hnxt = hbase + (cur ^ 1) * (MTILE * HID);

    for (int i = 0; i < TPW; ++i) {
      const int m0 = m0s[i], n0 = n0s[i];
      v8f ar, az, ain, ahn;
#pragma unroll
      for (int e = 0; e < 8; ++e) { ar[e] = brv[i]; az[e] = bzv[i]; ain[e] = binv[i]; ahn[e] = bhnv[i]; }

      // ---- fused input projection: gi tile, K = 128 over LDS-x / Wihb ----
      {
        const __bf16* ap = xcur + (long)(m0 + colLocal) * KP;   // A row (M = lane&15)
#pragma unroll
        for (int k0 = 0; k0 < KP; k0 += 32) {
          const int ka = k0 + (hl ? 16 : 0);      // A layout: 16 consecutive K
          v16bf a = load_frag(ap + ka, ap + ka + 8);
          const int kb = k0 + (hl ? 16 : 0);      // B layout: 16 consecutive K
          const __bf16* br_ = Wihb + (long)(n0 + colLocal) * KP + kb;
          const __bf16* bz_ = Wihb + (long)(HID + n0 + colLocal) * KP + kb;
          const __bf16* bn_ = Wihb + (long)(2 * HID + n0 + colLocal) * KP + kb;
          v16bf b0 = load_frag(br_, br_ + 8);
          v16bf b1 = load_frag(bz_, bz_ + 8);
          v16bf b2 = load_frag(bn_, bn_ + 8);
          ar  = __builtin_amdgcn_wmma_f32_16x16x32_bf16(false, a, false, b0, (short)0, ar,  false, false);
          az  = __builtin_amdgcn_wmma_f32_16x16x32_bf16(false, a, false, b1, (short)0, az,  false, false);
          ain = __builtin_amdgcn_wmma_f32_16x16x32_bf16(false, a, false, b2, (short)0, ain, false, false);
        }
      }
      // ---- hidden projection: gh tile, K = 512 over h(LDS) / Whhb ----
      {
        const __bf16* ap = hcur + (long)(m0 + colLocal) * HID;
#pragma unroll 4
        for (int k0 = 0; k0 < HID; k0 += 32) {
          const int ka = k0 + (hl ? 16 : 0);
          v16bf a = load_frag(ap + ka, ap + ka + 8);
          const int kb = k0 + (hl ? 16 : 0);
          const __bf16* br_ = Whhb + (long)(n0 + colLocal) * HID + kb;
          const __bf16* bz_ = Whhb + (long)(HID + n0 + colLocal) * HID + kb;
          const __bf16* bn_ = Whhb + (long)(2 * HID + n0 + colLocal) * HID + kb;
          v16bf b0 = load_frag(br_, br_ + 8);
          v16bf b1 = load_frag(bz_, bz_ + 8);
          v16bf b2 = load_frag(bn_, bn_ + 8);
          ar  = __builtin_amdgcn_wmma_f32_16x16x32_bf16(false, a, false, b0, (short)0, ar,  false, false);
          az  = __builtin_amdgcn_wmma_f32_16x16x32_bf16(false, a, false, b1, (short)0, az,  false, false);
          ahn = __builtin_amdgcn_wmma_f32_16x16x32_bf16(false, a, false, b2, (short)0, ahn, false, false);
        }
      }
      // ---- gates + h update directly on accumulator lanes ----
      // C/D layout: VGPR e, lane l -> M = e + 8*(l>=16), N = l&15
#pragma unroll
      for (int e = 0; e < 8; ++e) {
        int row = m0 + e + (hl << 3);
        int col = n0 + colLocal;
        float r = 1.0f / (1.0f + __expf(-ar[e]));
        float z = 1.0f / (1.0f + __expf(-az[e]));
        float n = tanhf(ain[e] + r * ahn[e]);
        float ho = (float)hcur[(long)row * HID + col];
        hnxt[(long)row * HID + col] = (__bf16)((1.0f - z) * n + z * ho);
      }
    }
    // Next-step slab DMA must be visible to all waves past the barrier.
    wait_async0();
    __syncthreads();
    cur ^= 1;
  }

  // ---- FC epilogue: out[row] = h . W_fc + b_fc (reuse dead h buffer) ----
  const __bf16* hfin = hbase + cur * (MTILE * HID);
  float* red = (float*)(hbase + (cur ^ 1) * (MTILE * HID));   // 32x8 floats
  {
    int r  = tid >> 3;      // 0..31 rows
    int c8 = tid & 7;
    float s = 0.0f;
    for (int k = c8; k < HID; k += 8)
      s += (float)hfin[(long)r * HID + k] * W_fc[k];
    red[r * 8 + c8] = s;
  }
  __syncthreads();
  if (tid < MTILE) {
    float s = 0.0f;
#pragma unroll
    for (int j = 0; j < 8; ++j) s += red[tid * 8 + j];
    out[blockBase + tid] = s + b_fc[0];
  }
}

// ---------------------------------------------------------------------------
extern "C" void kernel_launch(void* const* d_in, const int* in_sizes, int n_in,
                              void* d_out, int out_size, void* d_ws, size_t ws_size,
                              hipStream_t stream) {
  const int*   x_in = (const int*)  d_in[0];
  const float* emb  = (const float*)d_in[1];
  const float* W_ih = (const float*)d_in[2];
  const float* W_hh = (const float*)d_in[3];
  const float* b_ih = (const float*)d_in[4];
  const float* b_hh = (const float*)d_in[5];
  const float* W_fc = (const float*)d_in[6];
  const float* b_fc = (const float*)d_in[7];
  float* out = (float*)d_out;

  unsigned char* ws = (unsigned char*)d_ws;
  size_t xb_bytes  = (size_t)BB * TT * KP * sizeof(__bf16);   // 128 MB
  size_t wih_bytes = (size_t)H3 * KP * sizeof(__bf16);        // 384 KB
  __bf16* Xb   = (__bf16*)ws;
  __bf16* Wihb = (__bf16*)(ws + xb_bytes);
  __bf16* Whhb = (__bf16*)(ws + xb_bytes + wih_bytes);        // 1.5 MB

  (void)hipFuncSetAttribute((const void*)gru_persistent_kernel,
                            hipFuncAttributeMaxDynamicSharedMemorySize, SMEM_BYTES);

  long n_embed = (long)BB * TT * KP;                          // 67,108,864
  embed_pack_kernel<<<(int)(n_embed / 256), 256, 0, stream>>>(x_in, emb, Xb);
  pack_weights_kernel<<<1024, 256, 0, stream>>>(W_ih, W_hh, Wihb, Whhb);
  gru_persistent_kernel<<<NBLK, THREADS, SMEM_BYTES, stream>>>(Xb, Wihb, Whhb,
                                                               b_ih, b_hh, W_fc, b_fc, out);
}
